// Focal_Attention_53283364274501
// MI455X (gfx1250) — compile-verified, matrix operand order checked by reading, not run on hardware
//
#include <hip/hip_runtime.h>
#include <hip/hip_bf16.h>

// ---------------------------------------------------------------------------
// Problem constants (from reference): B=4, N=4096, M=1024, C=384, H=8, hd=48
// ---------------------------------------------------------------------------
#define B_   4
#define N_   4096
#define M_   1024
#define C_   384
#define H_   8
#define HD_  48
#define HDP_ 64     // head dim padded to 64 (two K=32 WMMA steps; pad is zero)
#define KV2_ 768    // 2*C for kv projection

typedef __attribute__((ext_vector_type(16))) __bf16         v16bf;
typedef __attribute__((ext_vector_type(8)))  float          v8f;
typedef __attribute__((ext_vector_type(8)))  unsigned short u16x8;

union BF16x16 {
    v16bf          v;
    u16x8          h[2];
    unsigned short s[16];
};

__device__ __forceinline__ unsigned short f2bf(float f) {
    union { float f; unsigned int u; } cv;
    cv.f = f;
    unsigned int u = cv.u;
    u += 0x7FFFu + ((u >> 16) & 1u);   // round-to-nearest-even
    return (unsigned short)(u >> 16);
}

#define WMMA_BF16(A, Bv, Cv) \
    __builtin_amdgcn_wmma_f32_16x16x32_bf16(false, (A), false, (Bv), (short)0, (Cv), false, false)

// ---------------------------------------------------------------------------
// Weight transpose + bf16 convert:  W[K, ncols] (row major) -> Wt[ncols, K]
// ---------------------------------------------------------------------------
__global__ void k_transpose_w(const float* __restrict__ W,
                              unsigned short* __restrict__ Wt,
                              int K, int ncols) {
    int idx = blockIdx.x * blockDim.x + threadIdx.x;
    if (idx >= K * ncols) return;
    int c = idx / K;
    int k = idx % K;
    Wt[idx] = f2bf(W[(size_t)k * ncols + c]);
}

// ---------------------------------------------------------------------------
// GEMM: Y[rows, cols] = X[rows, K] (f32 -> bf16) @ Wt[cols, K]
// One wave computes a 32x64 tile (2 row tiles x 4 col tiles, 8 accumulators):
// each A fragment feeds 4 WMMAs, each B fragment 2; all 4 B loads issue as one
// clause before the 8 back-to-back WMMAs.
// ---------------------------------------------------------------------------
__global__ __launch_bounds__(32) void k_gemm_f32a(
        const float* __restrict__ X, const unsigned short* __restrict__ Wt,
        float* __restrict__ Y, int rows, int K, int cols) {
    const int cblk = cols >> 6;                 // 64-wide column blocks
    const int rt   = blockIdx.x / cblk;         // 32-row tiles
    const int cb   = blockIdx.x % cblk;
    const int lane = threadIdx.x;
    const int half = lane >> 4;
    const int lo   = lane & 15;
    const int row0 = rt * 32 + lo;
    const int row1 = row0 + 16;

    v8f acc[2][4] = {};
    for (int k0 = 0; k0 < K; k0 += 32) {
        const int kb = k0 + half * 8;
        // B fragments first: one clause of 8 global_load_b128
        BF16x16 bfr[4];
        #pragma unroll
        for (int t = 0; t < 4; ++t) {
            const unsigned short* wp =
                Wt + ((size_t)(cb * 64 + t * 16 + lo)) * K + kb;
            bfr[t].h[0] = *(const u16x8*)(wp);
            bfr[t].h[1] = *(const u16x8*)(wp + 16);
        }
        // A fragments (f32 -> bf16 in registers)
        BF16x16 a0, a1;
        {
            const float* xp = X + (size_t)row0 * K + kb;
            #pragma unroll
            for (int i = 0; i < 8; ++i) {
                a0.s[i]     = f2bf(xp[i]);
                a0.s[8 + i] = f2bf(xp[16 + i]);
            }
            xp = X + (size_t)row1 * K + kb;
            #pragma unroll
            for (int i = 0; i < 8; ++i) {
                a1.s[i]     = f2bf(xp[i]);
                a1.s[8 + i] = f2bf(xp[16 + i]);
            }
        }
        // 8 back-to-back WMMAs
        #pragma unroll
        for (int t = 0; t < 4; ++t) {
            acc[0][t] = WMMA_BF16(a0.v, bfr[t].v, acc[0][t]);
            acc[1][t] = WMMA_BF16(a1.v, bfr[t].v, acc[1][t]);
        }
    }
    #pragma unroll
    for (int rb = 0; rb < 2; ++rb) {
        #pragma unroll
        for (int j = 0; j < 8; ++j) {
            int r = rt * 32 + rb * 16 + half * 8 + j;
            size_t base = (size_t)r * cols + cb * 64 + lo;
            Y[base]      = acc[rb][0][j];
            Y[base + 16] = acc[rb][1][j];
            Y[base + 32] = acc[rb][2][j];
            Y[base + 48] = acc[rb][3][j];
        }
    }
}

// ---------------------------------------------------------------------------
// Pack kernels: fp32 projection results -> WMMA-friendly bf16 layouts
// ---------------------------------------------------------------------------
__global__ void k_pack_q(const float* __restrict__ Qf, unsigned short* __restrict__ Qbf) {
    int idx = blockIdx.x * blockDim.x + threadIdx.x;   // (((b*H+h)*N+n)*64+d)
    int d    = idx & (HDP_ - 1);
    int rest = idx >> 6;
    int n    = rest % N_;
    int bh   = rest / N_;
    int h    = bh % H_;
    int b    = bh / H_;
    unsigned short v = 0;
    if (d < HD_) v = f2bf(Qf[((size_t)(b * N_ + n)) * C_ + h * HD_ + d]);
    Qbf[idx] = v;
}

__global__ void k_pack_k(const float* __restrict__ KVf, unsigned short* __restrict__ Kbf) {
    int idx = blockIdx.x * blockDim.x + threadIdx.x;   // (((b*H+h)*M+m)*64+d)
    int d    = idx & (HDP_ - 1);
    int rest = idx >> 6;
    int m    = rest % M_;
    int bh   = rest / M_;
    int h    = bh % H_;
    int b    = bh / H_;
    unsigned short v = 0;
    if (d < HD_) v = f2bf(KVf[((size_t)(b * M_ + m)) * KV2_ + h * HD_ + d]);
    Kbf[idx] = v;
}

__global__ void k_pack_v(const float* __restrict__ KVf, unsigned short* __restrict__ Vt) {
    int idx = blockIdx.x * blockDim.x + threadIdx.x;   // (((b*H+h)*48+d)*M+m)
    int m    = idx % M_;
    int rest = idx / M_;
    int d    = rest % HD_;
    int bh   = rest / HD_;
    int h    = bh % H_;
    int b    = bh / H_;
    Vt[idx] = f2bf(KVf[((size_t)(b * M_ + m)) * KV2_ + C_ + h * HD_ + d]);
}

// ---------------------------------------------------------------------------
// Flash attention: one wave per (b, h, 32-query tile), 32-key chunks.
// Two 16-query tiles share every K and V fragment.
// ---------------------------------------------------------------------------
__global__ __launch_bounds__(32) void k_attn_flash(
        const unsigned short* __restrict__ Qbf,
        const unsigned short* __restrict__ Kbf,
        const unsigned short* __restrict__ Vt,
        unsigned short* __restrict__ Xbf) {
    __shared__ alignas(16) unsigned short Pshr[2][16 * 32];

    const int NT   = N_ / 32;
    const int nt   = blockIdx.x % NT;
    const int bh   = blockIdx.x / NT;
    const int b    = bh / H_;
    const int h    = bh % H_;
    const int lane = threadIdx.x;
    const int half = lane >> 4;
    const int lo   = lane & 15;
    const float scale = 0.14433756729740643f;  // 48^-0.5

    // Q A-operands: 2 query tiles x 2 K-steps (padded head dim = 64)
    BF16x16 qa[2][2];
    #pragma unroll
    for (int q = 0; q < 2; ++q) {
        const unsigned short* qp =
            Qbf + (((size_t)bh * N_) + nt * 32 + q * 16 + lo) * HDP_ + half * 8;
        qa[q][0].h[0] = *(const u16x8*)(qp);
        qa[q][0].h[1] = *(const u16x8*)(qp + 16);
        qa[q][1].h[0] = *(const u16x8*)(qp + 32);
        qa[q][1].h[1] = *(const u16x8*)(qp + 48);
    }

    v8f o[2][3] = {};
    float mrow[2][8], lrow[2][8];
    #pragma unroll
    for (int q = 0; q < 2; ++q)
        #pragma unroll
        for (int j = 0; j < 8; ++j) { mrow[q][j] = -1e30f; lrow[q][j] = 0.0f; }

    for (int kc = 0; kc < M_; kc += 32) {
        // ---- K fragments for two 16-key sub-tiles (shared by both q tiles) ----
        BF16x16 bk[2][2];
        #pragma unroll
        for (int t = 0; t < 2; ++t) {
            const unsigned short* kp =
                Kbf + (((size_t)bh * M_) + kc + t * 16 + lo) * HDP_ + half * 8;
            bk[t][0].h[0] = *(const u16x8*)(kp);
            bk[t][0].h[1] = *(const u16x8*)(kp + 16);
            bk[t][1].h[0] = *(const u16x8*)(kp + 32);
            bk[t][1].h[1] = *(const u16x8*)(kp + 48);
        }
        // prefetch next chunk of K and V into WGP cache (global_prefetch_b8)
        if (kc + 32 < M_) {
            __builtin_prefetch(Kbf + (((size_t)bh * M_) + kc + 32 + lo) * HDP_, 0, 0);
            __builtin_prefetch(Vt + (((size_t)bh * HD_) + lo) * M_ + kc + 32, 0, 0);
        }

        // ---- scores: 2 q-tiles x 2 key-tiles, K dim = 64 (padded) ----
        v8f s[2][2];
        #pragma unroll
        for (int q = 0; q < 2; ++q)
            #pragma unroll
            for (int t = 0; t < 2; ++t) {
                v8f acc = {};
                acc = WMMA_BF16(qa[q][0].v, bk[t][0].v, acc);
                acc = WMMA_BF16(qa[q][1].v, bk[t][1].v, acc);
                s[q][t] = acc;
            }

        // ---- V fragments for this chunk: batch loads ahead of PV WMMAs ----
        BF16x16 bv[3];
        #pragma unroll
        for (int ct = 0; ct < 3; ++ct) {
            const unsigned short* vp =
                Vt + (((size_t)bh * HD_) + ct * 16 + lo) * M_ + kc + half * 8;
            bv[ct].h[0] = *(const u16x8*)(vp);
            bv[ct].h[1] = *(const u16x8*)(vp + 16);
        }

        // ---- online softmax per q tile ----
        float alpha[2][8];
        __syncthreads();
        #pragma unroll
        for (int q = 0; q < 2; ++q) {
            #pragma unroll
            for (int j = 0; j < 8; ++j) {
                float a0 = s[q][0][j] * scale;
                float a1 = s[q][1][j] * scale;
                float mx = fmaxf(a0, a1);
                #pragma unroll
                for (int msk = 1; msk < 16; msk <<= 1)
                    mx = fmaxf(mx, __shfl_xor(mx, msk, 32));
                float mnew = fmaxf(mrow[q][j], mx);
                alpha[q][j] = __expf(mrow[q][j] - mnew);
                float e0 = __expf(a0 - mnew);
                float e1 = __expf(a1 - mnew);
                float rs = e0 + e1;
                #pragma unroll
                for (int msk = 1; msk < 16; msk <<= 1)
                    rs += __shfl_xor(rs, msk, 32);
                lrow[q][j] = lrow[q][j] * alpha[q][j] + rs;
                mrow[q][j] = mnew;
                int r = half * 8 + j;
                Pshr[q][r * 32 + lo]      = f2bf(e0);
                Pshr[q][r * 32 + 16 + lo] = f2bf(e1);
            }
        }
        __syncthreads();

        // ---- P back as 16x32 A operands ----
        BF16x16 pa[2];
        #pragma unroll
        for (int q = 0; q < 2; ++q) {
            const unsigned short* pp = &Pshr[q][lo * 32 + half * 8];
            pa[q].h[0] = *(const u16x8*)(pp);
            pa[q].h[1] = *(const u16x8*)(pp + 16);
        }

        // ---- rescale running output, accumulate P @ V (6 back-to-back WMMAs) ----
        #pragma unroll
        for (int q = 0; q < 2; ++q)
            #pragma unroll
            for (int j = 0; j < 8; ++j) {
                o[q][0][j] *= alpha[q][j];
                o[q][1][j] *= alpha[q][j];
                o[q][2][j] *= alpha[q][j];
            }
        #pragma unroll
        for (int ct = 0; ct < 3; ++ct) {
            o[0][ct] = WMMA_BF16(pa[0].v, bv[ct].v, o[0][ct]);
            o[1][ct] = WMMA_BF16(pa[1].v, bv[ct].v, o[1][ct]);
        }
    }

    // ---- normalize, scatter into X[B,N,C] (bf16) ----
    #pragma unroll
    for (int q = 0; q < 2; ++q) {
        const size_t outbase =
            ((size_t)b * N_ + (size_t)nt * 32 + q * 16) * C_ + h * HD_;
        #pragma unroll
        for (int j = 0; j < 8; ++j) {
            float inv = 1.0f / lrow[q][j];
            int r = half * 8 + j;
            size_t ro = outbase + (size_t)r * C_;
            Xbf[ro + lo]      = f2bf(o[q][0][j] * inv);
            Xbf[ro + 16 + lo] = f2bf(o[q][1][j] * inv);
            Xbf[ro + 32 + lo] = f2bf(o[q][2][j] * inv);
        }
    }
}

// ---------------------------------------------------------------------------
// Output projection + bias + residual; 32x64 tile per wave like k_gemm_f32a
// ---------------------------------------------------------------------------
__global__ __launch_bounds__(32) void k_gemm_proj(
        const unsigned short* __restrict__ Xbf,
        const unsigned short* __restrict__ Wt,
        const float* __restrict__ bias,
        const float* __restrict__ Qres,
        float* __restrict__ out) {
    const int K = C_, cols = C_;
    const int cblk = cols >> 6;                 // 6 blocks of 64
    const int rt   = blockIdx.x / cblk;         // 32-row tiles
    const int cb   = blockIdx.x % cblk;
    const int lane = threadIdx.x;
    const int half = lane >> 4;
    const int lo   = lane & 15;
    const int row0 = rt * 32 + lo;
    const int row1 = row0 + 16;

    v8f acc[2][4] = {};
    for (int k0 = 0; k0 < K; k0 += 32) {
        const int kb = k0 + half * 8;
        BF16x16 bfr[4];
        #pragma unroll
        for (int t = 0; t < 4; ++t) {
            const unsigned short* wp =
                Wt + ((size_t)(cb * 64 + t * 16 + lo)) * K + kb;
            bfr[t].h[0] = *(const u16x8*)(wp);
            bfr[t].h[1] = *(const u16x8*)(wp + 16);
        }
        BF16x16 a0, a1;
        {
            const unsigned short* xp = Xbf + (size_t)row0 * K + kb;
            a0.h[0] = *(const u16x8*)(xp);
            a0.h[1] = *(const u16x8*)(xp + 16);
            xp = Xbf + (size_t)row1 * K + kb;
            a1.h[0] = *(const u16x8*)(xp);
            a1.h[1] = *(const u16x8*)(xp + 16);
        }
        #pragma unroll
        for (int t = 0; t < 4; ++t) {
            acc[0][t] = WMMA_BF16(a0.v, bfr[t].v, acc[0][t]);
            acc[1][t] = WMMA_BF16(a1.v, bfr[t].v, acc[1][t]);
        }
    }
    #pragma unroll
    for (int rb = 0; rb < 2; ++rb) {
        #pragma unroll
        for (int j = 0; j < 8; ++j) {
            int r = rt * 32 + rb * 16 + half * 8 + j;
            size_t base = (size_t)r * cols + cb * 64 + lo;
            out[base]      = acc[rb][0][j] + bias[cb * 64 + lo]      + Qres[base];
            out[base + 16] = acc[rb][1][j] + bias[cb * 64 + 16 + lo] + Qres[base + 16];
            out[base + 32] = acc[rb][2][j] + bias[cb * 64 + 32 + lo] + Qres[base + 32];
            out[base + 48] = acc[rb][3][j] + bias[cb * 64 + 48 + lo] + Qres[base + 48];
        }
    }
}

// ---------------------------------------------------------------------------
// Host launch
// ---------------------------------------------------------------------------
extern "C" void kernel_launch(void* const* d_in, const int* in_sizes, int n_in,
                              void* d_out, int out_size, void* d_ws, size_t ws_size,
                              hipStream_t stream) {
    const float* q_x   = (const float*)d_in[0];
    const float* kv_x  = (const float*)d_in[1];
    const float* Wq    = (const float*)d_in[2];
    const float* Wkv   = (const float*)d_in[3];
    const float* Wproj = (const float*)d_in[4];
    const float* bproj = (const float*)d_in[5];

    char* ws = (char*)d_ws;
    float*          Qf32   = (float*)(ws + 0);                 // 25165824 B
    float*          KVf32  = (float*)(ws + 25165824);          // 12582912 B
    unsigned short* Qbf    = (unsigned short*)(ws + 37748736); // 16777216 B
    unsigned short* Kbf    = (unsigned short*)(ws + 54525952); //  4194304 B
    unsigned short* Vt     = (unsigned short*)(ws + 58720256); //  3145728 B
    unsigned short* Xbf    = (unsigned short*)(ws + 61865984); // 12582912 B
    unsigned short* WqT    = (unsigned short*)(ws + 74448896); //   294912 B
    unsigned short* WkvT   = (unsigned short*)(ws + 74743808); //   589824 B
    unsigned short* WprojT = (unsigned short*)(ws + 75333632); //   294912 B

    k_transpose_w<<<(C_ * C_   + 255) / 256, 256, 0, stream>>>(Wq,    WqT,    C_, C_);
    k_transpose_w<<<(C_ * KV2_ + 255) / 256, 256, 0, stream>>>(Wkv,   WkvT,   C_, KV2_);
    k_transpose_w<<<(C_ * C_   + 255) / 256, 256, 0, stream>>>(Wproj, WprojT, C_, C_);

    k_gemm_f32a<<<(B_ * N_ / 32) * (C_ / 64),   32, 0, stream>>>(q_x,  WqT,  Qf32,  B_ * N_, C_, C_);
    k_gemm_f32a<<<(B_ * M_ / 32) * (KV2_ / 64), 32, 0, stream>>>(kv_x, WkvT, KVf32, B_ * M_, C_, KV2_);

    k_pack_q<<<(B_ * H_ * N_ * HDP_) / 256, 256, 0, stream>>>(Qf32,  Qbf);
    k_pack_k<<<(B_ * H_ * M_ * HDP_) / 256, 256, 0, stream>>>(KVf32, Kbf);
    k_pack_v<<<(B_ * H_ * HD_ * M_) / 256, 256, 0, stream>>>(KVf32, Vt);

    k_attn_flash<<<B_ * H_ * (N_ / 32), 32, 0, stream>>>(Qbf, Kbf, Vt, Xbf);

    k_gemm_proj<<<(B_ * N_ / 32) * (C_ / 64), 32, 0, stream>>>(
        Xbf, WprojT, bproj, Qf32, (float*)d_out);
}
